// MLPCritic_27230092657226
// MI455X (gfx1250) — compile-verified, matrix operand order checked by reading, not run on hardware
//
#include <hip/hip_runtime.h>
#include <hip/hip_bf16.h>

#define NN 4096
#define EE 8192
#define MM 6144
#define DD 64
#define HHID 256
#define IN_LEN (MM * (3 + DD))   // 411648

typedef __attribute__((ext_vector_type(16))) __bf16 v16bf;
typedef __attribute__((ext_vector_type(8)))  __bf16 v8bf;
typedef __attribute__((ext_vector_type(8)))  float  v8f;

__device__ __forceinline__ v8f zero8() {
    v8f z;
#pragma unroll
    for (int i = 0; i < 8; ++i) z[i] = 0.f;
    return z;
}

// Load a 16-bit A/B fragment: lane holds 16 bf16 K-values of a 32-wide K window
// from one row (contiguous along K). Per ISA 16-bit layout:
//   lanes 0-15 : K = {0..7, 16..23}; lanes 16-31 : K = {8..15, 24..31}
__device__ __forceinline__ v16bf load_frag_row(const __bf16* __restrict__ row,
                                               int k0, int klo) {
    v8bf lo = *(const v8bf*)(row + k0 + klo);
    v8bf hi = *(const v8bf*)(row + k0 + klo + 16);
    v16bf r;
#pragma unroll
    for (int i = 0; i < 8; ++i) { r[i] = lo[i]; r[i + 8] = hi[i]; }
    return r;
}

// Fused scaled-Gram kernel (register-blocked over 4 n-blocks):
//   OUT[mblk.., j] += sum_n ( ADJ[m,n] * sum_k A[m,k]*B[n,k] ) * Wsec[j,n]
// A,B  : bf16 row-major [dimOut x K] / [dimIn x K]
// ADJ  : f32 [dimOut x dimIn]
// Wsec : bf16 [16*DB x dimIn] (row j contiguous over n)
// OUT  : f32 [dimOut x 16*DB], atomically accumulated
template <int DB>
__global__ __launch_bounds__(256) void gram_fused(
    const __bf16* __restrict__ A, const __bf16* __restrict__ B,
    const float* __restrict__ ADJ, const __bf16* __restrict__ Wsec,
    float* __restrict__ OUT, int dimIn, int K, int nbPerGroup)
{
    const int lane = threadIdx.x & 31;
    const int w    = threadIdx.x >> 5;
    const int half = lane >> 4;
    const int l15  = lane & 15;
    const int klo  = half ? 8 : 0;

    const int mblk  = blockIdx.x * 16;
    const int group = blockIdx.y * (blockDim.x >> 5) + w;
    const int nb0   = group * nbPerGroup;

    __shared__ float Pt[8][16][17];

    const __bf16* arow = A + (size_t)(mblk + l15) * K;

    v8f acc[DB];
#pragma unroll
    for (int d = 0; d < DB; ++d) acc[d] = zero8();

    for (int nbi = 0; nbi < nbPerGroup; nbi += 4) {
        const __bf16* brow[4];
        v8f p[4];
#pragma unroll
        for (int j = 0; j < 4; ++j) {
            brow[j] = B + (size_t)(((nb0 + nbi + j) << 4) + l15) * K;
            p[j] = zero8();
        }

        // K-pass: one A fragment feeds 4 independent WMMA accumulator chains
        for (int e = 0; e < K; e += 32) {
            v16bf a = load_frag_row(arow, e, klo);
#pragma unroll
            for (int j = 0; j < 4; ++j) {
                v16bf b = load_frag_row(brow[j], e, klo);
                p[j] = __builtin_amdgcn_wmma_f32_16x16x32_bf16(
                           false, a, false, b, (short)0, p[j], false, false);
            }
        }

        // epilogue per n-block: adjacency mask, LDS transpose, stage-2 WMMA
#pragma unroll
        for (int j = 0; j < 4; ++j) {
            const int nblk = (nb0 + nbi + j) << 4;
            const int gn   = nblk + l15;
#pragma unroll
            for (int r = 0; r < 8; ++r)
                p[j][r] *= ADJ[(size_t)(mblk + half * 8 + r) * dimIn + gn];

            __syncthreads();
#pragma unroll
            for (int r = 0; r < 8; ++r)
                Pt[w][half * 8 + r][l15] = p[j][r];
            __syncthreads();

            v16bf ap;
#pragma unroll
            for (int i = 0; i < 8; ++i) {
                ap[i]     = (__bf16)Pt[w][l15][klo + i];
                ap[i + 8] = (__bf16)0.f;         // K 16..31 padded with zero
            }

#pragma unroll
            for (int d = 0; d < DB; ++d) {
                const __bf16* wrow = Wsec + (size_t)(d * 16 + l15) * dimIn + nblk;
                v8bf lo = *(const v8bf*)(wrow + klo);
                v16bf bw;
#pragma unroll
                for (int i = 0; i < 8; ++i) { bw[i] = lo[i]; bw[i + 8] = (__bf16)0.f; }
                acc[d] = __builtin_amdgcn_wmma_f32_16x16x32_bf16(
                             false, ap, false, bw, (short)0, acc[d], false, false);
            }
        }
    }

#pragma unroll
    for (int d = 0; d < DB; ++d)
#pragma unroll
        for (int r = 0; r < 8; ++r)
            atomicAdd(&OUT[(size_t)(mblk + half * 8 + r) * (16 * DB) + d * 16 + l15],
                      acc[d][r]);
}

// ---------------- elementwise / small kernels ----------------

__global__ void kzero(float* p, int n) {
    int i = blockIdx.x * blockDim.x + threadIdx.x;
    if (i < n) p[i] = 0.f;
}

__global__ void kconv(const float* __restrict__ src, __bf16* __restrict__ dst, size_t n) {
    size_t i = (size_t)blockIdx.x * blockDim.x + threadIdx.x;
    if (i < n) dst[i] = (__bf16)src[i];
}

// dst[i] = bf16(src[i] * s[i % rowlen])  (column scaling of row-major matrix)
__global__ void kscale_cols(const float* __restrict__ src, const float* __restrict__ s,
                            __bf16* __restrict__ dst, int rowlen, size_t n) {
    size_t i = (size_t)blockIdx.x * blockDim.x + threadIdx.x;
    if (i < n) dst[i] = (__bf16)(src[i] * s[i % rowlen]);
}

// TbT[e,n] = bf16(T[n,e]); TxT[e,n] = bf16(T[n,e] * x0[n])
__global__ void ktranspose(const float* __restrict__ T, const float* __restrict__ x0,
                           __bf16* __restrict__ TbT, __bf16* __restrict__ TxT) {
    __shared__ float tile[32][33];
    const int e0 = blockIdx.x * 32, n0 = blockIdx.y * 32;
    const int tx = threadIdx.x, ty = threadIdx.y;
    for (int r = ty; r < 32; r += 8)
        tile[r][tx] = T[(size_t)(n0 + r) * EE + e0 + tx];   // tile[a][b]=T[n0+a][e0+b]
    __syncthreads();
    for (int r = ty; r < 32; r += 8) {
        float v  = tile[tx][r];                              // T[n0+tx][e0+r]
        float xv = x0[n0 + tx];
        size_t o = (size_t)(e0 + r) * NN + n0 + tx;
        TbT[o] = (__bf16)v;
        TxT[o] = (__bf16)(v * xv);
    }
}

// out[j*dim + n] = bf16(x[n] * wrow[j]) ; j in [0,16)
__global__ void kouter(const float* __restrict__ x, const float* __restrict__ wrow,
                       __bf16* __restrict__ out, int dim) {
    int i = blockIdx.x * blockDim.x + threadIdx.x;
    int j = i / dim, n = i - j * dim;
    out[i] = (__bf16)(x[n] * wrow[j]);
}

__global__ void kz1(const float* __restrict__ He, float* __restrict__ z1) {
    int e = blockIdx.x * blockDim.x + threadIdx.x;
    float s = 0.f;
#pragma unroll
    for (int j = 0; j < 16; ++j) s += fmaxf(He[e * 16 + j], 0.f);
    z1[e] = s * (1.f / 16.f);
}

// Gvb[j*N + n] = bf16( sum_k relu(Hv[n,k]) * Wv2[k, j] ), j in [0,64)
__global__ void kGvb(const float* __restrict__ Hv, const float* __restrict__ Wv2,
                     __bf16* __restrict__ Gvb) {
    int i = blockIdx.x * blockDim.x + threadIdx.x;
    int j = i >> 12;          // / NN
    int n = i & (NN - 1);
    float s = 0.f;
#pragma unroll
    for (int k = 0; k < 16; ++k) s += fmaxf(Hv[n * 16 + k], 0.f) * Wv2[k * 64 + j];
    Gvb[i] = (__bf16)s;
}

// inclusive prefix scan of keep=flag>0 over M=6144 (1024 threads x 6 elems)
__global__ void kscan(const float* __restrict__ flag, int* __restrict__ idx) {
    __shared__ int part[1024];
    const int t = threadIdx.x, base = t * 6;
    int loc[6], s = 0;
#pragma unroll
    for (int k = 0; k < 6; ++k) { loc[k] = (flag[base + k] > 0.f) ? 1 : 0; s += loc[k]; }
    part[t] = s;
    __syncthreads();
    for (int off = 1; off < 1024; off <<= 1) {
        int v = part[t];
        int add = (t >= off) ? part[t - off] : 0;
        __syncthreads();
        part[t] = v + add;
        __syncthreads();
    }
    int pre = (t == 0) ? 0 : part[t - 1];
#pragma unroll
    for (int k = 0; k < 6; ++k) {
        pre += loc[k];
        int v = pre - 1;
        v = v < 0 ? 0 : (v > NN - 1 ? NN - 1 : v);
        idx[base + k] = v;
    }
}

__global__ void klse(const float* __restrict__ c, float* __restrict__ out) {
    __shared__ float red[1024];
    const int t = threadIdx.x;
    float m = -1e30f;
    for (int i = t; i < MM; i += 1024) m = fmaxf(m, c[i]);
    red[t] = m; __syncthreads();
    for (int off = 512; off; off >>= 1) { if (t < off) red[t] = fmaxf(red[t], red[t + off]); __syncthreads(); }
    float mx = red[0]; __syncthreads();
    float s = 0.f;
    for (int i = t; i < MM; i += 1024) s += __expf(c[i] - mx);
    red[t] = s; __syncthreads();
    for (int off = 512; off; off >>= 1) { if (t < off) red[t] += red[t + off]; __syncthreads(); }
    if (t == 0) out[0] = mx + __logf(red[0]);
}

__global__ void kobs(const float* __restrict__ cost, const float* __restrict__ flag,
                     const float* __restrict__ feat, const float* __restrict__ emb,
                     const int* __restrict__ idx, const float* __restrict__ lse,
                     float* __restrict__ obs) {
    int o = blockIdx.x * blockDim.x + threadIdx.x;
    if (o >= IN_LEN) return;
    int i = o / 67;
    int c = o - i * 67;
    float v;
    if      (c == 0) v = cost[i] - lse[0];
    else if (c == 1) v = flag[i];
    else if (c == 2) v = feat[i];
    else             v = (flag[i] > 0.f) ? emb[(size_t)idx[i] * 64 + (c - 3)] : 0.f;
    obs[o] = v;
}

// h1[h] += sum over 256-row chunk of obs[i]*W1[i,h]  (bandwidth-bound, 421 MB of W1)
__global__ __launch_bounds__(256) void kgemv(const float* __restrict__ obs,
                                             const float* __restrict__ W1,
                                             float* __restrict__ h1) {
    __shared__ float ob[256];
    const int t = threadIdx.x;
    const size_t base = (size_t)blockIdx.x * 256;
    ob[t] = obs[base + t];
    __syncthreads();
    float acc = 0.f;
    for (int i = 0; i < 256; ++i)
        acc += ob[i] * W1[(base + i) * HHID + t];
    atomicAdd(&h1[t], acc);
}

__global__ void kfinal(const float* __restrict__ h1, const float* __restrict__ b1,
                       const float* __restrict__ W2, const float* __restrict__ b2,
                       const float* __restrict__ W3, const float* __restrict__ b3,
                       float* __restrict__ out) {
    __shared__ float hA[256], hB[256];
    const int t = threadIdx.x;
    hA[t] = tanhf(h1[t] + b1[t]);
    __syncthreads();
    float s = 0.f;
    for (int k = 0; k < 256; ++k) s += hA[k] * W2[k * 256 + t];
    hB[t] = tanhf(s + b2[t]) * W3[t];
    __syncthreads();
    for (int off = 128; off; off >>= 1) { if (t < off) hB[t] += hB[t + off]; __syncthreads(); }
    if (t == 0) out[0] = hB[0] + b3[0];
}

extern "C" void kernel_launch(void* const* d_in, const int* in_sizes, int n_in,
                              void* d_out, int out_size, void* d_ws, size_t ws_size,
                              hipStream_t stream) {
    const float* X      = (const float*)d_in[0];
    const float* Z      = (const float*)d_in[1];
    const float* adj_e  = (const float*)d_in[2];
    const float* adj_v  = (const float*)d_in[3];
    const float* T      = (const float*)d_in[4];
    const float* cost   = (const float*)d_in[5];
    const float* flag   = (const float*)d_in[6];
    const float* feat   = (const float*)d_in[7];
    const float* Wv1    = (const float*)d_in[8];
    const float* We1    = (const float*)d_in[9];
    const float* Wv2    = (const float*)d_in[10];
    const float* W1     = (const float*)d_in[11];
    const float* b1     = (const float*)d_in[12];
    const float* W2     = (const float*)d_in[13];
    const float* b2     = (const float*)d_in[14];
    const float* W3     = (const float*)d_in[15];
    const float* b3     = (const float*)d_in[16];
    float* out = (float*)d_out;

    char* ws = (char*)d_ws;
    size_t off = 0;
    auto alloc = [&](size_t bytes) -> void* {
        off = (off + 255) & ~(size_t)255;
        void* p = ws + off;
        off += bytes;
        return p;
    };

    const size_t TE = (size_t)NN * EE;
    __bf16* Tb  = (__bf16*)alloc(TE * 2);          // bf16(T)           [N x E]
    __bf16* TbT = (__bf16*)alloc(TE * 2);          // bf16(T^T)         [E x N]
    __bf16* TxT = (__bf16*)alloc(TE * 2);          // bf16(T^T * x0)    [E x N]
    __bf16* Tz  = (__bf16*)alloc(TE * 2);          // bf16(T * z)       [N x E]
    __bf16* Uvb = (__bf16*)alloc((size_t)16 * NN * 2);
    __bf16* Ueb = (__bf16*)alloc((size_t)16 * EE * 2);
    __bf16* Gvb = (__bf16*)alloc((size_t)64 * NN * 2);
    float*  Hv  = (float*)alloc((size_t)NN * 16 * 4);
    float*  He  = (float*)alloc((size_t)EE * 16 * 4);
    float*  z1  = (float*)alloc((size_t)EE * 4);
    float*  emb = (float*)alloc((size_t)NN * 64 * 4);
    int*    idx = (int*)alloc((size_t)MM * 4);
    float*  obs = (float*)alloc((size_t)IN_LEN * 4);
    float*  h1  = (float*)alloc(256 * 4);
    float*  lse = (float*)alloc(16);

    // --- preprocessing / precision conversion ---
    kconv      <<<TE / 256, 256, 0, stream>>>(T, Tb, TE);
    kscale_cols<<<TE / 256, 256, 0, stream>>>(T, Z, Tz, EE, TE);       // T * z0
    ktranspose <<<dim3(EE / 32, NN / 32), dim3(32, 8), 0, stream>>>(T, X, TbT, TxT);
    kouter     <<<(16 * NN) / 256, 256, 0, stream>>>(X, Wv1, Uvb, NN); // Uv = X@Wv1
    kouter     <<<(16 * EE) / 256, 256, 0, stream>>>(Z, We1, Ueb, EE); // Ue = Z@We1
    kzero<<<(NN * 16) / 256, 256, 0, stream>>>(Hv, NN * 16);
    kzero<<<(EE * 16) / 256, 256, 0, stream>>>(He, EE * 16);
    kzero<<<(NN * 64) / 256, 256, 0, stream>>>(emb, NN * 64);
    kzero<<<1, 256, 0, stream>>>(h1, 256);

    // --- the three fused Gram einsums (WMMA bf16) ---
    // Hv_pre = (adj_v .* ((T z0) T^T)) @ Uv        [N x 16]
    gram_fused<1><<<dim3(NN / 16, 4), 256, 0, stream>>>(Tz, Tb, adj_v, Uvb, Hv, NN, EE, 8);
    // He_pre = (adj_e .* ((T^T x0) T)) @ Ue        [E x 16]  (E x E never materialized)
    gram_fused<1><<<dim3(EE / 16, 8), 256, 0, stream>>>(TxT, TbT, adj_e, Ueb, He, EE, NN, 8);
    kz1<<<EE / 256, 256, 0, stream>>>(He, z1);                          // z1 = mean relu
    kscale_cols<<<TE / 256, 256, 0, stream>>>(T, z1, Tz, EE, TE);       // Tz <- T * z1
    kGvb<<<(64 * NN) / 256, 256, 0, stream>>>(Hv, Wv2, Gvb);            // relu(Hv)@Wv2
    // emb = (adj_v .* ((T z1) T^T)) @ Gv           [N x 64]
    gram_fused<4><<<dim3(NN / 16, 4), 256, 0, stream>>>(Tz, Tb, adj_v, Gvb, emb, NN, EE, 8);

    // --- gather / observation assembly / MLP head ---
    kscan<<<1, 1024, 0, stream>>>(flag, idx);
    klse <<<1, 1024, 0, stream>>>(cost, lse);
    kobs <<<(IN_LEN + 255) / 256, 256, 0, stream>>>(cost, flag, feat, emb, idx, lse, obs);
    kgemv<<<IN_LEN / 256, 256, 0, stream>>>(obs, W1, h1);
    kfinal<<<1, 256, 0, stream>>>(h1, b1, W2, b2, W3, b3, out);
}
